// AST_LSTM_86595130622200
// MI455X (gfx1250) — matrix operand order repaired and checked by reading, not compiled, not run on hardware
//
#include <hip/hip_runtime.h>
#include <math.h>

// ---------------------------------------------------------------------------
// AST-GGNN (GatedGraphConv + GRUCell), 15 leaf-peeling iterations.
// N = 30000 nodes, D = 384, gates = 3*D = 1152.
// Core GEMMs run on V_WMMA_F32_16X16X32_F16 (f16 in, f32 accumulate).
// Algebraic fusion: mean-agg is linear, so  m_agg @ w_ih^T ==
// (mean-agg hidden) @ (conv_W @ w_ih^T); Wc precomputed once per launch.
// ---------------------------------------------------------------------------

#define D_EMB 384
#define G3D   1152   // 3 * D_EMB
#define ITERS 15

typedef _Float16 v8h  __attribute__((ext_vector_type(8)));
typedef _Float16 v16h __attribute__((ext_vector_type(16)));
typedef float    v8f  __attribute__((ext_vector_type(8)));

// ---------------------------------------------------------------------------
// once-per-launch kernels
// ---------------------------------------------------------------------------

__global__ void init_kernel(const float* __restrict__ V, float* __restrict__ hidden,
                            _Float16* __restrict__ x_h0, int* __restrict__ active,
                            int ND, int M) {
    int t = blockIdx.x * blockDim.x + threadIdx.x;
    if (t < ND) {
        float v = V[t];
        hidden[t] = v;
        x_h0[t]   = (_Float16)v;
    }
    if (t < M) active[t] = 1;
}

// wcT[c][k] = sum_j conv[k][j] * w_ih[c][j]   (== (conv @ w_ih^T) transposed)
// whh_h = f16(w_hh)   both stored [1152][384] row-major f16
__global__ void prep_weights_kernel(const float* __restrict__ convW,
                                    const float* __restrict__ w_ih,
                                    const float* __restrict__ w_hh,
                                    _Float16* __restrict__ wcT,
                                    _Float16* __restrict__ whh_h, int total) {
    int t = blockIdx.x * blockDim.x + threadIdx.x;
    if (t >= total) return;
    int c = t / D_EMB, k = t % D_EMB;
    const float* ck = convW + (size_t)k * D_EMB;
    const float* wc = w_ih  + (size_t)c * D_EMB;
    float s = 0.f;
    for (int j = 0; j < D_EMB; ++j) s += ck[j] * wc[j];
    wcT[t]   = (_Float16)s;
    whh_h[t] = (_Float16)w_hh[t];
}

// ---------------------------------------------------------------------------
// per-iteration kernels
// ---------------------------------------------------------------------------

__global__ void zero_kernel(float4* __restrict__ h_agg4, int* __restrict__ tgt,
                            int* __restrict__ cnt, int* __restrict__ nu,
                            int* __restrict__ flag, int nd4, int n) {
    int t = blockIdx.x * blockDim.x + threadIdx.x;
    if (t < nd4) h_agg4[t] = make_float4(0.f, 0.f, 0.f, 0.f);
    if (t < n) { tgt[t] = 0; cnt[t] = 0; nu[t] = 0; }
    if (t == 0) *flag = 0;
}

__global__ void count_kernel(const int* __restrict__ active, const int* __restrict__ dstE,
                             int* __restrict__ tgt, int* __restrict__ flag, int M) {
    int e = blockIdx.x * blockDim.x + threadIdx.x;
    if (e >= M) return;
    if (active[e]) { atomicAdd(&tgt[dstE[e]], 1); *flag = 1; }
}

__global__ void use_kernel(int* __restrict__ active, int* __restrict__ useE,
                           const int* __restrict__ srcE, const int* __restrict__ dstE,
                           const int* __restrict__ tgt, int* __restrict__ nu, int M) {
    int e = blockIdx.x * blockDim.x + threadIdx.x;
    if (e >= M) return;
    int u = (active[e] && tgt[srcE[e]] == 0) ? 1 : 0;
    useE[e] = u;
    if (u) {
        active[e] = 0;
        nu[srcE[e]] = 1;   // benign races: all writers store 1
        nu[dstE[e]] = 1;
    }
}

// index_map = cumsum(node_used) - 1, single 1024-thread workgroup (N <= 32768)
__global__ void scan_kernel(const int* __restrict__ nu, int* __restrict__ im, int n) {
    __shared__ int sums[1024];
    int t = threadIdx.x;
    int start = t * 32;
    int local[32];
    int s = 0;
#pragma unroll
    for (int i = 0; i < 32; ++i) {
        int idx = start + i;
        int v = (idx < n) ? nu[idx] : 0;
        s += v;
        local[i] = s;
    }
    sums[t] = s;
    __syncthreads();
    for (int ofs = 1; ofs < 1024; ofs <<= 1) {
        int v = 0;
        if (t >= ofs) v = sums[t - ofs];
        __syncthreads();
        sums[t] += v;
        __syncthreads();
    }
    int base = sums[t] - s;   // exclusive prefix of this chunk
#pragma unroll
    for (int i = 0; i < 32; ++i) {
        int idx = start + i;
        if (idx < n) im[idx] = base + local[i] - 1;
    }
}

// h_agg[index_map[dst]] += hidden[index_map[src]] ; cnt[index_map[dst]] += 1
// (matches the reference's compacted-index quirk exactly)
__global__ void scatter_kernel(const int* __restrict__ useE, const int* __restrict__ srcE,
                               const int* __restrict__ dstE, const int* __restrict__ im,
                               const float* __restrict__ hidden, float* __restrict__ h_agg,
                               int* __restrict__ cnt, int M) {
    int t = blockIdx.x * blockDim.x + threadIdx.x;
    int e = t / 96;                // 96 chunks of 4 floats = 384
    if (e >= M) return;
    if (!useE[e]) return;
    int r = t % 96;
    int k = r * 4;
    int s2 = im[srcE[e]];
    int t2 = im[dstE[e]];
    const float* sp = hidden + (size_t)s2 * D_EMB + k;
    float* dp = h_agg + (size_t)t2 * D_EMB + k;
    atomicAdd(dp + 0, sp[0]);
    atomicAdd(dp + 1, sp[1]);
    atomicAdd(dp + 2, sp[2]);
    atomicAdd(dp + 3, sp[3]);
    if (r == 0) atomicAdd(&cnt[t2], 1);
}

// a_h = f16(h_agg / max(cnt,1))
__global__ void convert_kernel(const float* __restrict__ h_agg, const int* __restrict__ cnt,
                               _Float16* __restrict__ a_h, int ND) {
    int t = blockIdx.x * blockDim.x + threadIdx.x;
    int i = t * 4;
    if (i >= ND) return;
    int row = i / D_EMB;           // all 4 elements in the same row (384 % 4 == 0)
    float inv = 1.0f / fmaxf((float)cnt[row], 1.0f);
#pragma unroll
    for (int u = 0; u < 4; ++u) a_h[i + u] = (_Float16)(h_agg[i + u] * inv);
}

// ---------------------------------------------------------------------------
// Fused dual-GEMM + GRU gate kernel.
// gi = a_h @ Wc(+b_ih), gh = x_h @ w_hh^T(+b_hh), both via WMMA f16->f32,
// gates + blend fused into the epilogue; f32 state updated in place, and the
// f16 next-state written into the ping-pong buffer for the next iteration.
//
// Operand layouts (ISA 7.12.2/7.12.4):
//   A 16x32 f16: lane<16 holds K {0..7,16..23}; lane>=16 holds K {8..15,24..31}
//   B 32x16 f16: lane -> output column; lane<16 holds K {0..7,16..23}; lane>=16 K {8..15,24..31}
//   C/D 16x16 f32: lane -> column, VGPR v -> row v (+8 for lanes>=16)
// ---------------------------------------------------------------------------

__device__ inline v16h load_a(const _Float16* rowk, int lane) {
    const _Float16* q = rowk + ((lane & 16) ? 8 : 0);
    v8h lo = *reinterpret_cast<const v8h*>(q);
    v8h hi = *reinterpret_cast<const v8h*>(q + 16);
    return __builtin_shufflevector(lo, hi, 0, 1, 2, 3, 4, 5, 6, 7,
                                   8, 9, 10, 11, 12, 13, 14, 15);
}

__device__ inline v16h load_b(const _Float16* rowk, int lane) {
    const _Float16* q = rowk + ((lane & 16) ? 8 : 0);
    v8h lo = *reinterpret_cast<const v8h*>(q);
    v8h hi = *reinterpret_cast<const v8h*>(q + 16);
    return __builtin_shufflevector(lo, hi, 0, 1, 2, 3, 4, 5, 6, 7,
                                   8, 9, 10, 11, 12, 13, 14, 15);
}

#define WMMA_F16(A, B, C) \
    __builtin_amdgcn_wmma_f32_16x16x32_f16(false, (A), false, (B), (short)0, (C), false, false)

__global__ __launch_bounds__(256) void gemm_gru_kernel(
    const _Float16* __restrict__ x_h, _Float16* __restrict__ x_next,
    const _Float16* __restrict__ a_h,
    const _Float16* __restrict__ wcT, const _Float16* __restrict__ whh_h,
    const float* __restrict__ b_ih, const float* __restrict__ b_hh,
    float* __restrict__ hidden, const int* __restrict__ flag, int N) {
    const int tid  = threadIdx.x;
    const int lane = tid & 31;
    const int wave = tid >> 5;
    const int l15  = lane & 15;

    const int ft = blockIdx.y * 8 + wave;   // feature tile 0..23
    const int fb = ft * 16;
    const int nb = blockIdx.x * 16;

    int arow = nb + l15;
    if (arow >= N) arow = N - 1;            // clamp (cndmask, EXEC stays full)
    const _Float16* ax_p = x_h + (size_t)arow * D_EMB;
    const _Float16* aa_p = a_h + (size_t)arow * D_EMB;

    const int col = fb + l15;               // output column for this lane
    const _Float16* bir_p = wcT   + (size_t)(col)          * D_EMB;
    const _Float16* biz_p = wcT   + (size_t)(D_EMB + col)  * D_EMB;
    const _Float16* bin_p = wcT   + (size_t)(2*D_EMB + col)* D_EMB;
    const _Float16* bhr_p = whh_h + (size_t)(col)          * D_EMB;
    const _Float16* bhz_p = whh_h + (size_t)(D_EMB + col)  * D_EMB;
    const _Float16* bhn_p = whh_h + (size_t)(2*D_EMB + col)* D_EMB;

    v8f q_ir = {}, q_iz = {}, q_in = {}, q_hr = {}, q_hz = {}, q_hn = {};

    for (int kb = 0; kb < D_EMB; kb += 32) {
        v16h aa = load_a(aa_p + kb, lane);
        v16h ax = load_a(ax_p + kb, lane);
        v16h b0 = load_b(bir_p + kb, lane);
        v16h b1 = load_b(biz_p + kb, lane);
        v16h b2 = load_b(bin_p + kb, lane);
        q_ir = WMMA_F16(aa, b0, q_ir);
        q_iz = WMMA_F16(aa, b1, q_iz);
        q_in = WMMA_F16(aa, b2, q_in);
        b0 = load_b(bhr_p + kb, lane);
        b1 = load_b(bhz_p + kb, lane);
        b2 = load_b(bhn_p + kb, lane);
        q_hr = WMMA_F16(ax, b0, q_hr);
        q_hz = WMMA_F16(ax, b1, q_hz);
        q_hn = WMMA_F16(ax, b2, q_hn);
    }

    // ------- GRU epilogue: C/D layout -> one column per lane, rows in VGPRs
    const int doW = flag[0];
    const float bir = b_ih[col],            bhr = b_hh[col];
    const float biz = b_ih[D_EMB + col],    bhz = b_hh[D_EMB + col];
    const float bin = b_ih[2*D_EMB + col],  bhn = b_hh[2*D_EMB + col];
    const int rbase = nb + ((lane & 16) ? 8 : 0);
#pragma unroll
    for (int v = 0; v < 8; ++v) {
        int row = rbase + v;
        if (row < N) {
            size_t idx = (size_t)row * D_EMB + col;
            float h  = hidden[idx];
            float r  = 1.f / (1.f + __expf(-(q_ir[v] + bir + q_hr[v] + bhr)));
            float z  = 1.f / (1.f + __expf(-(q_iz[v] + biz + q_hz[v] + bhz)));
            float nn = tanhf(q_in[v] + bin + r * (q_hn[v] + bhn));
            float hn = (1.f - z) * nn + z * h;
            float out = doW ? hn : h;
            if (doW) hidden[idx] = hn;
            x_next[idx] = (_Float16)out;   // f16 state for next iteration's A
        }
    }
}

// ---------------------------------------------------------------------------
// host launcher
// ---------------------------------------------------------------------------

extern "C" void kernel_launch(void* const* d_in, const int* in_sizes, int n_in,
                              void* d_out, int out_size, void* d_ws, size_t ws_size,
                              hipStream_t stream) {
    (void)n_in; (void)out_size; (void)ws_size;
    const float* V     = (const float*)d_in[0];
    const int*   E     = (const int*)d_in[1];
    const float* convW = (const float*)d_in[2];
    const float* w_ih  = (const float*)d_in[3];
    const float* w_hh  = (const float*)d_in[4];
    const float* b_ih  = (const float*)d_in[5];
    const float* b_hh  = (const float*)d_in[6];
    float* hidden = (float*)d_out;

    const int ND = in_sizes[0];
    const int N  = ND / D_EMB;
    const int M  = in_sizes[1] / 2;
    const int* srcE = E;
    const int* dstE = E + M;

    // ---- carve workspace (256B aligned slices)
    char* p = (char*)d_ws;
    auto carve = [&](size_t bytes) -> void* {
        void* r = (void*)p;
        p += (bytes + 255) & ~(size_t)255;
        return r;
    };
    _Float16* x_h0  = (_Float16*)carve((size_t)ND * 2);
    _Float16* x_h1  = (_Float16*)carve((size_t)ND * 2);
    _Float16* a_h   = (_Float16*)carve((size_t)ND * 2);
    float*    h_agg = (float*)   carve((size_t)ND * 4);
    _Float16* wcT   = (_Float16*)carve((size_t)G3D * D_EMB * 2);
    _Float16* whh_h = (_Float16*)carve((size_t)G3D * D_EMB * 2);
    int* tgt_cnt    = (int*)carve((size_t)N * 4);
    int* cnt        = (int*)carve((size_t)N * 4);
    int* node_used  = (int*)carve((size_t)N * 4);
    int* index_map  = (int*)carve((size_t)N * 4);
    int* active     = (int*)carve((size_t)M * 4);
    int* useE       = (int*)carve((size_t)M * 4);
    int* flag       = (int*)carve(4);

    const int B = 256;
    int mx = ND > M ? ND : M;
    init_kernel<<<(mx + B - 1) / B, B, 0, stream>>>(V, hidden, x_h0, active, ND, M);

    const int wtot = G3D * D_EMB;
    prep_weights_kernel<<<(wtot + B - 1) / B, B, 0, stream>>>(convW, w_ih, w_hh, wcT, whh_h, wtot);

    const int nd4 = ND / 4;
    const dim3 ggrid((N + 15) / 16, 3);

    for (int it = 0; it < ITERS; ++it) {
        _Float16* x_cur  = (it & 1) ? x_h1 : x_h0;
        _Float16* x_nxt  = (it & 1) ? x_h0 : x_h1;
        zero_kernel<<<(nd4 + B - 1) / B, B, 0, stream>>>((float4*)h_agg, tgt_cnt, cnt,
                                                         node_used, flag, nd4, N);
        count_kernel<<<(M + B - 1) / B, B, 0, stream>>>(active, dstE, tgt_cnt, flag, M);
        use_kernel<<<(M + B - 1) / B, B, 0, stream>>>(active, useE, srcE, dstE,
                                                      tgt_cnt, node_used, M);
        scan_kernel<<<1, 1024, 0, stream>>>(node_used, index_map, N);
        long long st = (long long)M * 96;
        scatter_kernel<<<(int)((st + B - 1) / B), B, 0, stream>>>(useE, srcE, dstE, index_map,
                                                                  hidden, h_agg, cnt, M);
        convert_kernel<<<(nd4 + B - 1) / B, B, 0, stream>>>(h_agg, cnt, a_h, ND);
        gemm_gru_kernel<<<ggrid, B, 0, stream>>>(x_cur, x_nxt, a_h, wcT, whh_h,
                                                 b_ih, b_hh, hidden, flag, N);
    }
}
